// WindowAttention_64381559767606
// MI455X (gfx1250) — matrix operand order repaired and checked by reading, not compiled, not run on hardware
//
#include <hip/hip_runtime.h>
#include <hip/hip_bf16.h>

typedef _Float16 v16h __attribute__((ext_vector_type(16)));
typedef float    v8f  __attribute__((ext_vector_type(8)));

namespace {

constexpr int kN    = 64;      // tokens per window (8x8)
constexpr int kD    = 384;
constexpr int kH    = 12;
constexpr int kHD   = 32;
constexpr int kD3   = 3 * kD;  // 1152
constexpr int kNW   = 64;
constexpr int kBias = 15 * 15; // (2*8-1)^2 = 225 rel-pos entries

constexpr int LDQ   = kD3 + 8; // halves
constexpr int LDX   = kD  + 8; // halves
constexpr int LDS_S = kN  + 4; // floats
constexpr int LDP   = kN  + 8; // halves
constexpr int LDC   = kD  + 8; // halves
constexpr int LDKT  = kN  + 8; // halves (72)
constexpr int LDM   = kN  + 4; // floats (mask)

constexpr int SQKV_BYTES = kN * LDQ * 2;                         // 148480
constexpr int SS_BYTES   = kN * LDS_S * 4;                       // 17408
constexpr int SP_BYTES   = kN * LDP * 2;                         // 9216
constexpr int SCTX_BYTES = kN * LDC * 2;                         // 50176
constexpr int REGB_BYTES = SS_BYTES + SP_BYTES + SCTX_BYTES;     // 76800 (>= x-stage 50176)
constexpr int SKT_BYTES  = kD * LDKT * 2;                        // 55296 (K^T, all 12 heads)
constexpr int SBIAS_BYTES= kBias * kH * 4;                       // 10800
constexpr int SMASK_BYTES= kN * LDM * 4;                         // 17408
constexpr int SMEM_BYTES = SQKV_BYTES + REGB_BYTES + SKT_BYTES + SBIAS_BYTES + SMASK_BYTES; // 308784 < 320KB

__device__ __forceinline__ v8f wmma16x16x32(v16h a, v16h b, v8f c) {
  return __builtin_amdgcn_wmma_f32_16x16x32_f16(false, a, false, b, (short)0, c, false, false);
}

// A fragment 16x32 f16: lane = M (mod 16); lanes 0-15 hold K {0..7,16..23}, lanes 16-31 hold K {8..15,24..31}.
__device__ __forceinline__ v16h load_a_frag(const _Float16* base, int ld, int row0, int col0) {
  const int lane = threadIdx.x & 31;
  const int m  = row0 + (lane & 15);
  const int kb = col0 + ((lane & 16) ? 8 : 0);
  v16h a;
#pragma unroll
  for (int j = 0; j < 8; ++j) {
    const int k = kb + 2 * j + ((j >= 4) ? 8 : 0);
    a[2 * j]     = base[m * ld + k];
    a[2 * j + 1] = base[m * ld + k + 1];
  }
  return a;
}

// B fragment 32x16 f16, memory row-major B[k][n] at base[k*ld + n]: lane = N (mod 16); lanes 0-15 hold K {0..7,16..23}, lanes 16-31 hold K {8..15,24..31}.
__device__ __forceinline__ v16h load_b_rowmajor(const _Float16* base, int ld, int krow0, int ncol0) {
  const int lane = threadIdx.x & 31;
  const _Float16* col = base + (size_t)krow0 * ld + ncol0 + (lane & 15);
  const int kb = (lane & 16) ? 8 : 0;
  v16h b;
#pragma unroll
  for (int j = 0; j < 8; ++j) {
    const int k = kb + 2 * j + ((j >= 4) ? 8 : 0);
    b[2 * j]     = col[(size_t)k * ld];
    b[2 * j + 1] = col[(size_t)(k + 1) * ld];
  }
  return b;
}

// D-tile (f32 C/D layout: VGPR r = row, lane&15 = col, lane>=16 -> rows 8..15) -> qkv LDS (+K^T mirror)
__device__ __forceinline__ void store_qkv_tile(_Float16* sQKV, _Float16* sKT,
                                               const float* __restrict__ b_qkv,
                                               v8f c, int row0, int col0) {
  const int lane   = threadIdx.x & 31;
  const int ncol   = col0 + (lane & 15);
  const float bias = b_qkv[ncol];
  const int rbase  = row0 + ((lane & 16) ? 8 : 0);
  const bool is_k  = (ncol >= kD) && (ncol < 2 * kD);   // uniform per tile
#pragma unroll
  for (int r = 0; r < 8; ++r) {
    const _Float16 hv = (_Float16)(c[r] + bias);
    sQKV[(rbase + r) * LDQ + ncol] = hv;
    if (is_k) sKT[(ncol - kD) * LDKT + (rbase + r)] = hv; // K^T: [h*32+d][token]
  }
}

__device__ __forceinline__ void store_out_tile(float* __restrict__ ob,
                                               const float* __restrict__ b_out,
                                               v8f c, int row0, int col0) {
  const int lane   = threadIdx.x & 31;
  const int ncol   = col0 + (lane & 15);
  const float bias = b_out[ncol];
  const int rbase  = row0 + ((lane & 16) ? 8 : 0);
#pragma unroll
  for (int r = 0; r < 8; ++r)
    ob[(rbase + r) * kD + ncol] = c[r] + bias;
}

__global__ void convert_weights_f16(const float* __restrict__ Wq, const float* __restrict__ Wo,
                                    _Float16* __restrict__ Wq_h, _Float16* __restrict__ Wo_h) {
  const int i = blockIdx.x * blockDim.x + threadIdx.x;
  if (i < kD * kD3) Wq_h[i] = (_Float16)Wq[i];
  if (i < kD * kD)  Wo_h[i] = (_Float16)Wo[i];
}

__global__ __launch_bounds__(256, 1)
void swin_window_attention_fused(const float* __restrict__ x,
                                 const float* __restrict__ attn_mask,
                                 const float* __restrict__ b_qkv,
                                 const float* __restrict__ b_out,
                                 const float* __restrict__ bias_table,
                                 const _Float16* __restrict__ Wq_h,
                                 const _Float16* __restrict__ Wo_h,
                                 float* __restrict__ out) {
  __shared__ alignas(16) unsigned char smem[SMEM_BYTES];
  _Float16* sQKV = (_Float16*)smem;                                      // [64][LDQ]
  _Float16* sX   = (_Float16*)(smem + SQKV_BYTES);                       // [64][LDX]   (phase 1 only)
  float*    sS   = (float*)   (smem + SQKV_BYTES);                       // [64][LDS_S] (phase 2)
  _Float16* sP   = (_Float16*)(smem + SQKV_BYTES + SS_BYTES);            // [64][LDP]
  _Float16* sCtx = (_Float16*)(smem + SQKV_BYTES + SS_BYTES + SP_BYTES); // [64][LDC]
  _Float16* sKT  = (_Float16*)(smem + SQKV_BYTES + REGB_BYTES);          // [384][LDKT]  K^T per head
  float*    sBias= (float*)   (smem + SQKV_BYTES + REGB_BYTES + SKT_BYTES);              // [225][12]
  float*    sMask= (float*)   (smem + SQKV_BYTES + REGB_BYTES + SKT_BYTES + SBIAS_BYTES);// [64][LDM]

  const int tid  = threadIdx.x;
  const int lane = tid & 31;
  const int wave = tid >> 5;
  const int b    = blockIdx.x;

  // ---- Phase 1a: stage x[b] as f16 + bias table + this window's mask (all LDS-resident) ----
  {
    const float4* xb4 = (const float4*)(x + (size_t)b * kN * kD);
#pragma unroll 4
    for (int idx = tid; idx < kN * kD / 4; idx += 256) {
      const float4 v = xb4[idx];
      const int e = idx * 4;
      _Float16* p = &sX[(e / kD) * LDX + (e % kD)];
      p[0] = (_Float16)v.x; p[1] = (_Float16)v.y; p[2] = (_Float16)v.z; p[3] = (_Float16)v.w;
    }
    for (int idx = tid; idx < kBias * kH; idx += 256)
      sBias[idx] = bias_table[idx];
    const float4* mb4 = (const float4*)(attn_mask + (size_t)(b & (kNW - 1)) * kN * kN);
#pragma unroll
    for (int idx = tid; idx < kN * kN / 4; idx += 256) {
      const float4 v = mb4[idx];
      const int e = idx * 4;
      float* p = &sMask[(e >> 6) * LDM + (e & 63)];
      p[0] = v.x; p[1] = v.y; p[2] = v.z; p[3] = v.w;
    }
  }
  __syncthreads();

  // ---- Phase 1b: qkv = x @ W_qkv + b_qkv (64x1152, K=384), 2x2 register-blocked ----
  // 288 tiles = (2 mi-pairs) x (36 ni-pairs) = 72 groups of 4 tiles
  for (int g = wave; g < 72; g += 8) {
    const int mi0 = (g & 1) * 32;
    const int ni0 = (g >> 1) * 32;
    v8f c00 = {}, c01 = {}, c10 = {}, c11 = {};
#pragma unroll
    for (int kt = 0; kt < 12; ++kt) {
      v16h a0 = load_a_frag(sX, LDX, mi0,      kt * 32);
      v16h a1 = load_a_frag(sX, LDX, mi0 + 16, kt * 32);
      v16h b0 = load_b_rowmajor(Wq_h, kD3, kt * 32, ni0);
      v16h b1 = load_b_rowmajor(Wq_h, kD3, kt * 32, ni0 + 16);
      c00 = wmma16x16x32(a0, b0, c00);
      c01 = wmma16x16x32(a0, b1, c01);
      c10 = wmma16x16x32(a1, b0, c10);
      c11 = wmma16x16x32(a1, b1, c11);
    }
    store_qkv_tile(sQKV, sKT, b_qkv, c00, mi0,      ni0);
    store_qkv_tile(sQKV, sKT, b_qkv, c01, mi0,      ni0 + 16);
    store_qkv_tile(sQKV, sKT, b_qkv, c10, mi0 + 16, ni0);
    store_qkv_tile(sQKV, sKT, b_qkv, c11, mi0 + 16, ni0 + 16);
  }
  __syncthreads();

  // ---- Phase 2: per-head attention ----
  const float scale = 0.17677669529663687f; // 1/sqrt(32)
  for (int h = 0; h < kH; ++h) {
    const int qoff = h * kHD;
    const int voff = 2 * kD + h * kHD;

    // S = Q K^T : 4x4 tiles, K = 32 (one WMMA each); B from contiguous K^T rows
    for (int t = wave; t < 16; t += 8) {
      const int mi = t & 3, ni = t >> 2;
      v16h a  = load_a_frag(sQKV, LDQ, mi * 16, qoff);
      v16h bb = load_b_rowmajor(sKT + (size_t)h * kHD * LDKT, LDKT, 0, ni * 16);
      v8f c = {};
      c = wmma16x16x32(a, bb, c);
      const int ncol  = ni * 16 + (lane & 15);
      const int rbase = mi * 16 + ((lane & 16) ? 8 : 0);
#pragma unroll
      for (int r = 0; r < 8; ++r)
        sS[(rbase + r) * LDS_S + ncol] = c[r];
    }
    __syncthreads();

    // softmax: 4 threads per row, 16 columns each; shfl_xor reductions over lanes {1,2}
    {
      const int i  = tid >> 2;       // row (token)
      const int j0 = (tid & 3) * 16; // quarter-row start
      const int r1 = i >> 3, c1 = i & 7;
      const float* mrow = sMask + i * LDM;
      float mx = -3.0e38f;
#pragma unroll
      for (int jj = 0; jj < 16; ++jj) {
        const int j  = j0 + jj;
        const int r2 = j >> 3, c2 = j & 7;
        const int ridx = (r1 - r2 + 7) * 15 + (c1 - c2 + 7);
        float l = sS[i * LDS_S + j] * scale + sBias[ridx * kH + h] + mrow[j];
        sS[i * LDS_S + j] = l;
        mx = fmaxf(mx, l);
      }
      mx = fmaxf(mx, __shfl_xor(mx, 1, 32));
      mx = fmaxf(mx, __shfl_xor(mx, 2, 32));
      float sum = 0.f;
#pragma unroll
      for (int jj = 0; jj < 16; ++jj) {
        const int j = j0 + jj;
        const float e = __expf(sS[i * LDS_S + j] - mx);
        sS[i * LDS_S + j] = e;
        sum += e;
      }
      sum += __shfl_xor(sum, 1, 32);
      sum += __shfl_xor(sum, 2, 32);
      const float inv = __frcp_rn(sum);
#pragma unroll
      for (int jj = 0; jj < 16; ++jj) {
        const int j = j0 + jj;
        sP[i * LDP + j] = (_Float16)(sS[i * LDS_S + j] * inv);
      }
    }
    __syncthreads();

    // ctx_h = P @ V : one 16x16 tile per wave, K = 64 (two WMMA)
    {
      const int mi = wave & 3, di = wave >> 2;
      v8f c = {};
#pragma unroll
      for (int s = 0; s < 2; ++s) {
        v16h a  = load_a_frag(sP, LDP, mi * 16, s * 32);
        v16h bb = load_b_rowmajor(sQKV + voff, LDQ, s * 32, di * 16);
        c = wmma16x16x32(a, bb, c);
      }
      const int dcol  = di * 16 + (lane & 15);
      const int rbase = mi * 16 + ((lane & 16) ? 8 : 0);
#pragma unroll
      for (int r = 0; r < 8; ++r)
        sCtx[(rbase + r) * LDC + h * kHD + dcol] = (_Float16)c[r];
    }
    __syncthreads();
  }

  // ---- Phase 3: out = ctx @ W_out + b_out (64x384, K=384), 2x2 register-blocked, f32 to HBM ----
  // 96 tiles = (2 mi-pairs) x (12 ni-pairs) = 24 groups of 4 tiles
  float* ob = out + (size_t)b * kN * kD;
  for (int g = wave; g < 24; g += 8) {
    const int mi0 = (g & 1) * 32;
    const int ni0 = (g >> 1) * 32;
    v8f c00 = {}, c01 = {}, c10 = {}, c11 = {};
#pragma unroll
    for (int kt = 0; kt < 12; ++kt) {
      v16h a0 = load_a_frag(sCtx, LDC, mi0,      kt * 32);
      v16h a1 = load_a_frag(sCtx, LDC, mi0 + 16, kt * 32);
      v16h b0 = load_b_rowmajor(Wo_h, kD, kt * 32, ni0);
      v16h b1 = load_b_rowmajor(Wo_h, kD, kt * 32, ni0 + 16);
      c00 = wmma16x16x32(a0, b0, c00);
      c01 = wmma16x16x32(a0, b1, c01);
      c10 = wmma16x16x32(a1, b0, c10);
      c11 = wmma16x16x32(a1, b1, c11);
    }
    store_out_tile(ob, b_out, c00, mi0,      ni0);
    store_out_tile(ob, b_out, c01, mi0,      ni0 + 16);
    store_out_tile(ob, b_out, c10, mi0 + 16, ni0);
    store_out_tile(ob, b_out, c11, mi0 + 16, ni0 + 16);
  }
}

} // namespace

extern "C" void kernel_launch(void* const* d_in, const int* in_sizes, int n_in,
                              void* d_out, int out_size, void* d_ws, size_t ws_size,
                              hipStream_t stream) {
  (void)in_sizes; (void)n_in; (void)out_size; (void)ws_size;
  const float* x          = (const float*)d_in[0];
  const float* attn_mask  = (const float*)d_in[1];
  const float* W_qkv      = (const float*)d_in[2];
  const float* b_qkv      = (const float*)d_in[3];
  const float* W_out      = (const float*)d_in[4];
  const float* b_out      = (const float*)d_in[5];
  const float* bias_table = (const float*)d_in[6];
  float* out = (float*)d_out;

  _Float16* Wq_h = (_Float16*)d_ws;                                 // 384*1152 halves
  _Float16* Wo_h = (_Float16*)((char*)d_ws + (size_t)kD * kD3 * 2); // 384*384 halves

  const int nconv = kD * kD3;
  convert_weights_f16<<<(nconv + 255) / 256, 256, 0, stream>>>(W_qkv, W_out, Wq_h, Wo_h);

  swin_window_attention_fused<<<2048, 256, 0, stream>>>(
      x, attn_mask, b_qkv, b_out, bias_table, Wq_h, Wo_h, out);
}
